// SDIM_7395933683868
// MI455X (gfx1250) — matrix operand order repaired and checked by reading, not compiled, not run on hardware
//
#include <hip/hip_runtime.h>
#include <hip/hip_bf16.h>
#include <math.h>

// ---------------- CDNA5 WMMA types / helpers ----------------
typedef __bf16   v16bf __attribute__((ext_vector_type(16)));
typedef float    v8f   __attribute__((ext_vector_type(8)));
typedef unsigned v4u   __attribute__((ext_vector_type(4)));

__device__ __forceinline__ __bf16 f2bf(float f) {
    unsigned u = __builtin_bit_cast(unsigned, f);
    unsigned r = u + 0x7FFFu + ((u >> 16) & 1u);   // round-to-nearest-even
    unsigned short h = (unsigned short)(r >> 16);
    return __builtin_bit_cast(__bf16, h);
}
// pack two floats into one dword holding two bf16 (lo = f0, hi = f1)
__device__ __forceinline__ unsigned packbf2(float f0, float f1) {
    unsigned u0 = __builtin_bit_cast(unsigned, f0);
    unsigned u1 = __builtin_bit_cast(unsigned, f1);
    unsigned r0 = u0 + 0x7FFFu + ((u0 >> 16) & 1u);
    unsigned r1 = u1 + 0x7FFFu + ((u1 >> 16) & 1u);
    return (r0 >> 16) | (r1 & 0xFFFF0000u);
}
__device__ __forceinline__ float bflo(unsigned u) { return __builtin_bit_cast(float, u << 16); }
__device__ __forceinline__ float bfhi(unsigned u) { return __builtin_bit_cast(float, u & 0xFFFF0000u); }

__device__ __forceinline__ v8f wmma_bf16(v16bf a, v16bf b, v8f c) {
    return __builtin_amdgcn_wmma_f32_16x16x32_bf16(
        false, a, false, b, (short)0, c, false, false);
}

// A fragment from bf16 NHWC: two contiguous 8-channel runs, masked (no branches).
__device__ __forceinline__ v16bf load_afrag(const __bf16* __restrict__ pix,
                                            int cc, int half, unsigned msk) {
    v4u a0 = *(const v4u*)(pix + cc + half * 8);
    v4u a1 = *(const v4u*)(pix + cc + 16 + half * 8);
    union { v4u u[2]; v16bf v; } r;
    r.u[0] = a0 & msk;
    r.u[1] = a1 & msk;
    return r.v;
}

// B fragment: pre-swizzled per-lane contiguous 32 bytes -> one vector load.
__device__ __forceinline__ v16bf load_bfrag(const __bf16* __restrict__ Wf,
                                            int chunk, int NF, int nf, int lane) {
    return *(const v16bf*)(Wf + (size_t)(((chunk * NF + nf) * 32 + lane) * 16));
}

__device__ __forceinline__ int iclamp(int v, int lo, int hi) {
    return v < lo ? lo : (v > hi ? hi : v);
}

// 13 non-zero taps of the folded ELK 5x5 kernel (dy,dx)
__constant__ int cDY[13] = {-1,-1,-1, 0, 0, 0, 1, 1, 1,  0, 0,-2, 2};
__constant__ int cDX[13] = {-1, 0, 1,-1, 0, 1,-1, 0, 1, -2, 2, 0, 0};

#define WW 128
#define HW 16384
#define NC 64

// ============ activation layout conversion: NCHW f32 -> NHWC bf16 ============
__global__ void k_x2bf(const float* __restrict__ x, __bf16* __restrict__ xbf) {
    __shared__ float tile[64][65];
    const int b = blockIdx.x >> 8;             // 2048 blocks = 8 * 256
    const int pix0 = (blockIdx.x & 255) * 64;
    const int t = threadIdx.x;                 // 256
    const float* xb = x + (size_t)b * NC * HW + pix0;
#pragma unroll
    for (int j = 0; j < 16; ++j) {
        int idx = j * 256 + t;                 // 4096 = 64c * 64p
        int c = idx >> 6, p = idx & 63;
        tile[c][p] = xb[(size_t)c * HW + p];
    }
    __syncthreads();
    __bf16* ob = xbf + ((size_t)b * HW + pix0) * 64;
#pragma unroll
    for (int j = 0; j < 16; ++j) {
        int idx = j * 256 + t;
        int p = idx >> 6, c = idx & 63;
        ob[(size_t)p * 64 + c] = f2bf(tile[c][p]);
    }
}

// 2x2 maxpool fused with NCHW->NHWC bf16: h1 [8,32,128,128] -> h1pb [8,64*64,32]
__global__ void k_pool_cvt(const float* __restrict__ h1, __bf16* __restrict__ out) {
    __shared__ float tile[32][257];
    const int b = blockIdx.x >> 6;             // 512 = 8*64
    const int yo = blockIdx.x & 63;
    const int t = threadIdx.x;                 // 256
    const float* hb = h1 + (size_t)b * 32 * HW + (yo * 2) * WW;
#pragma unroll
    for (int j = 0; j < 32; ++j) {
        int idx = j * 256 + t;                 // 8192 = 32c * 256(2rows x 128)
        int c = idx >> 8, r = idx & 255;
        tile[c][r] = hb[(size_t)c * HW + (r >> 7) * WW + (r & 127)];
    }
    __syncthreads();
    __bf16* ob = out + ((size_t)b * 4096 + yo * 64) * 32;
#pragma unroll
    for (int j = 0; j < 8; ++j) {
        int idx = j * 256 + t;                 // 2048 = 64xo * 32c
        int c = idx & 31, xo = idx >> 5;
        float v = fmaxf(fmaxf(tile[c][xo * 2], tile[c][xo * 2 + 1]),
                        fmaxf(tile[c][128 + xo * 2], tile[c][128 + xo * 2 + 1]));
        ob[(size_t)xo * 32 + c] = f2bf(v);
    }
}

// ============ weight repack: f32 -> bf16, pre-swizzled fragment layout ============
__global__ void k_cvt_stn1(const float* __restrict__ w, __bf16* __restrict__ out) {
    int i = blockIdx.x * 256 + threadIdx.x;    // 98 chunks * 2 nf * 32 * 16 = 100352
    if (i >= 100352) return;
    int e = i & 15, lane = (i >> 4) & 31, nf = (i >> 9) & 1, chunk = i >> 10;
    int k = chunk * 32 + (e & 8) * 2 + (lane >> 4) * 8 + (e & 7);
    int col = nf * 16 + (lane & 15);
    int c = k & 63, tap = k >> 6, ty = tap / 7, tx = tap % 7;
    out[i] = f2bf(w[((col * 64 + c) * 7 + ty) * 7 + tx]);
}

__global__ void k_cvt_stn2(const float* __restrict__ w, __bf16* __restrict__ out) {
    int i = blockIdx.x * 256 + threadIdx.x;    // 25 chunks * 4 * 32 * 16 = 51200
    if (i >= 51200) return;
    int e = i & 15, lane = (i >> 4) & 31, nf = (i >> 9) & 3, chunk = i >> 11;
    int k = chunk * 32 + (e & 8) * 2 + (lane >> 4) * 8 + (e & 7);
    int col = nf * 16 + (lane & 15);
    int c = k & 31, tap = k >> 5, ty = tap / 5, tx = tap % 5;
    out[i] = f2bf(w[((col * 32 + c) * 5 + ty) * 5 + tx]);
}

__global__ void k_cvt_dcn(const float* __restrict__ w, __bf16* __restrict__ out) {
    int i = blockIdx.x * 256 + threadIdx.x;    // 18 chunks * 4 * 32 * 16 = 36864
    if (i >= 36864) return;
    int e = i & 15, lane = (i >> 4) & 31, nf = (i >> 9) & 3, chunk = i >> 11;
    int k = chunk * 32 + (e & 8) * 2 + (lane >> 4) * 8 + (e & 7);
    int col = nf * 16 + (lane & 15);
    int c = k & 63, k9 = k >> 6;
    out[i] = f2bf(w[(col * 64 + c) * 9 + k9]);
}

// Fold ELK (3 branches + 1x1) into one 13-tap conv, emitted in fragment layout.
__global__ void k_elk_fold(const float* __restrict__ w3,  const float* __restrict__ w15,
                           const float* __restrict__ w51, const float* __restrict__ w1,
                           const float* __restrict__ b3,  const float* __restrict__ b15,
                           const float* __restrict__ b51, const float* __restrict__ b1,
                           __bf16* __restrict__ Weff, float* __restrict__ beff) {
    int i = blockIdx.x * 256 + threadIdx.x;    // 26 chunks * 4 * 32 * 16 = 53248 (+64 bias)
    if (i < 53248) {
        int e = i & 15, lane = (i >> 4) & 31, nf = (i >> 9) & 3, chunk = i >> 11;
        int k = chunk * 32 + (e & 8) * 2 + (lane >> 4) * 8 + (e & 7);
        int o = nf * 16 + (lane & 15);
        int t = k >> 6, c = k & 63;
        int dy = cDY[t], dx = cDX[t];
        float s = 0.f;
        if (dy >= -1 && dy <= 1 && dx >= -1 && dx <= 1) {
            int kk = (dy + 1) * 3 + (dx + 1);
            for (int m = 0; m < 64; ++m)
                s += w1[o * 192 + m] * w3[(m * 64 + c) * 9 + kk];
        }
        if (dy == 0) {
            int kx = dx + 2;
            for (int m = 0; m < 64; ++m)
                s += w1[o * 192 + 64 + m] * w15[(m * 64 + c) * 5 + kx];
        }
        if (dx == 0) {
            int ky = dy + 2;
            for (int m = 0; m < 64; ++m)
                s += w1[o * 192 + 128 + m] * w51[(m * 64 + c) * 5 + ky];
        }
        Weff[i] = f2bf(s);
    } else if (i < 53248 + 64) {
        int o = i - 53248;
        float s = b1[o];
        for (int m = 0; m < 64; ++m) {
            s += w1[o * 192 + m]       * b3[m];
            s += w1[o * 192 + 64 + m]  * b15[m];
            s += w1[o * 192 + 128 + m] * b51[m];
        }
        beff[o] = s;
    }
}

// ============ STN conv1: 7x7, 64->32, pad 3, ReLU (WMMA, bf16 NHWC in) ============
__global__ void k_stn_conv1(const __bf16* __restrict__ xbf, const __bf16* __restrict__ Wb,
                            const float* __restrict__ bias, float* __restrict__ out) {
    const int tile = blockIdx.x, lane = threadIdx.x;   // 8192 x 32
    const int px0 = (tile & 7) * 16, y = (tile >> 3) & 127, b = tile >> 10;
    const int m = lane & 15, half = lane >> 4;
    const int xc = px0 + m;
    const __bf16* xb = xbf + (size_t)b * HW * 64;

    v8f acc0 = {}, acc1 = {};
    int chunk = 0;                                     // 2 chunks per tap, taps in dy/dx order
    for (int dy = -3; dy <= 3; ++dy) {
        const int yy = y + dy;
        const int yc = iclamp(yy, 0, 127);
        const unsigned mrow = ((unsigned)yy < 128u) ? ~0u : 0u;
        for (int dx = -3; dx <= 3; ++dx) {
            const int xx = xc + dx;
            const unsigned msk = mrow & (((unsigned)xx < 128u) ? ~0u : 0u);
            const __bf16* pix = xb + (size_t)(yc * WW + iclamp(xx, 0, 127)) * 64;
            __builtin_prefetch(pix + 64 * WW, 0, 1);
#pragma unroll
            for (int cc = 0; cc < 64; cc += 32) {
                v16bf a = load_afrag(pix, cc, half, msk);
                acc0 = wmma_bf16(a, load_bfrag(Wb, chunk, 2, 0, lane), acc0);
                acc1 = wmma_bf16(a, load_bfrag(Wb, chunk, 2, 1, lane), acc1);
                ++chunk;
            }
        }
    }
#pragma unroll
    for (int nf = 0; nf < 2; ++nf) {
        int co = nf * 16 + m;
        float bi = bias[co];
        float* op = out + ((size_t)b * 32 + co) * HW + y * WW + px0 + half * 8;
        v8f acc = nf ? acc1 : acc0;
#pragma unroll
        for (int i = 0; i < 8; ++i) op[i] = fmaxf(acc[i] + bi, 0.f);
    }
}

// ============ STN conv2: 5x5, 32->64, pad 2, ReLU, 64x64 (WMMA) ============
__global__ void k_stn_conv2(const __bf16* __restrict__ in, const __bf16* __restrict__ Wb,
                            const float* __restrict__ bias, float* __restrict__ out) {
    const int tile = blockIdx.x, lane = threadIdx.x;   // 2048 x 32
    const int px0 = (tile & 3) * 16, y = (tile >> 2) & 63, b = tile >> 8;
    const int m = lane & 15, half = lane >> 4;
    const int xc = px0 + m;
    const __bf16* xb = in + (size_t)b * 4096 * 32;

    v8f acc[4] = {};
    int tap = 0;
    for (int dy = -2; dy <= 2; ++dy) {
        const int yy = y + dy;
        const int yc = iclamp(yy, 0, 63);
        const unsigned mrow = ((unsigned)yy < 64u) ? ~0u : 0u;
        for (int dx = -2; dx <= 2; ++dx, ++tap) {
            const int xx = xc + dx;
            const unsigned msk = mrow & (((unsigned)xx < 64u) ? ~0u : 0u);
            const __bf16* pix = xb + (size_t)(yc * 64 + iclamp(xx, 0, 63)) * 32;
            v16bf a = load_afrag(pix, 0, half, msk);   // Cin=32 -> one chunk per tap
#pragma unroll
            for (int nf = 0; nf < 4; ++nf)
                acc[nf] = wmma_bf16(a, load_bfrag(Wb, tap, 4, nf, lane), acc[nf]);
        }
    }
#pragma unroll
    for (int nf = 0; nf < 4; ++nf) {
        int co = nf * 16 + m;
        float bi = bias[co];
        float* op = out + ((size_t)b * 64 + co) * 4096 + y * 64 + px0 + half * 8;
#pragma unroll
        for (int i = 0; i < 8; ++i) op[i] = fmaxf(acc[nf][i] + bi, 0.f);
    }
}

// ============ global average pool: h2 [8,64,64,64] -> feat [8,64] ============
__global__ void k_gap(const float* __restrict__ h2, float* __restrict__ feat) {
    __shared__ float red[256];
    const int ch = blockIdx.x;                          // 512
    const float* p = h2 + (size_t)ch * 4096;
    float s = 0.f;
    for (int i = threadIdx.x; i < 4096; i += 256) s += p[i];
    red[threadIdx.x] = s;
    __syncthreads();
    for (int st = 128; st > 0; st >>= 1) {
        if (threadIdx.x < st) red[threadIdx.x] += red[threadIdx.x + st];
        __syncthreads();
    }
    if (threadIdx.x == 0) feat[ch] = red[0] * (1.f / 4096.f);
}

__global__ void k_fc(const float* __restrict__ feat, const float* __restrict__ fc_w,
                     const float* __restrict__ fc_b, float* __restrict__ theta) {
    int t = threadIdx.x;
    if (t >= 48) return;
    int b = t / 6, j = t % 6;
    float s = fc_b[j];
    for (int k = 0; k < 64; ++k) s += feat[b * 64 + k] * fc_w[j * 64 + k];
    theta[t] = s;
}

// ============ STN bilinear warp (from bf16 NHWC) -> Fs f32 NCHW ============
__global__ void k_sample(const __bf16* __restrict__ xbf, const float* __restrict__ theta,
                         float* __restrict__ Fs) {
    int p = blockIdx.x * 256 + threadIdx.x;             // 131072
    int xc = p & 127, y = (p >> 7) & 127, b = p >> 14;
    const float* th = theta + b * 6;
    float Xn = (2.f * xc + 1.f) * (1.f / 128.f) - 1.f;
    float Yn = (2.f * y  + 1.f) * (1.f / 128.f) - 1.f;
    float gx = th[0] * Xn + th[1] * Yn + th[2];
    float gy = th[3] * Xn + th[4] * Yn + th[5];
    float ix = ((gx + 1.f) * 128.f - 1.f) * 0.5f;
    float iy = ((gy + 1.f) * 128.f - 1.f) * 0.5f;

    float fy = floorf(iy), fx = floorf(ix);
    float wy1 = iy - fy, wy0 = 1.f - wy1;
    float wx1 = ix - fx, wx0 = 1.f - wx1;
    int y0 = (int)fy, x0 = (int)fx, y1 = y0 + 1, x1 = x0 + 1;
    bool vy0 = (unsigned)y0 < 128u, vy1 = (unsigned)y1 < 128u;
    bool vx0 = (unsigned)x0 < 128u, vx1 = (unsigned)x1 < 128u;
    int cy0 = iclamp(y0, 0, 127), cy1 = iclamp(y1, 0, 127);
    int cx0 = iclamp(x0, 0, 127), cx1 = iclamp(x1, 0, 127);
    float w00 = wy0 * wx0 * (float)(vy0 && vx0);
    float w01 = wy0 * wx1 * (float)(vy0 && vx1);
    float w10 = wy1 * wx0 * (float)(vy1 && vx0);
    float w11 = wy1 * wx1 * (float)(vy1 && vx1);
    const __bf16* xb = xbf + (size_t)b * HW * 64;
    const __bf16* p00 = xb + (size_t)(cy0 * WW + cx0) * 64;
    const __bf16* p01 = xb + (size_t)(cy0 * WW + cx1) * 64;
    const __bf16* p10 = xb + (size_t)(cy1 * WW + cx0) * 64;
    const __bf16* p11 = xb + (size_t)(cy1 * WW + cx1) * 64;

    float* fb = Fs + (size_t)b * NC * HW + y * WW + xc;
    for (int cb = 0; cb < 64; cb += 8) {
        v4u q00 = *(const v4u*)(p00 + cb);
        v4u q01 = *(const v4u*)(p01 + cb);
        v4u q10 = *(const v4u*)(p10 + cb);
        v4u q11 = *(const v4u*)(p11 + cb);
#pragma unroll
        for (int j = 0; j < 4; ++j) {
            float gl = w00 * bflo(q00[j]) + w01 * bflo(q01[j]) +
                       w10 * bflo(q10[j]) + w11 * bflo(q11[j]);
            float gh = w00 * bfhi(q00[j]) + w01 * bfhi(q01[j]) +
                       w10 * bfhi(q10[j]) + w11 * bfhi(q11[j]);
            fb[(size_t)(cb + 2 * j) * HW]     = gl;
            fb[(size_t)(cb + 2 * j + 1) * HW] = gh;
        }
    }
}

// ============ Deformable conv: WMMA GEMM, bilinear A from bf16 NHWC ============
__global__ void k_dcn(const __bf16* __restrict__ xbf, const float* __restrict__ off,
                      const __bf16* __restrict__ Wd, const float* __restrict__ dbias,
                      float* __restrict__ Fd) {
    const int tile = blockIdx.x, lane = threadIdx.x;    // 8192 x 32
    const int px0 = (tile & 7) * 16, y = (tile >> 3) & 127, b = tile >> 10;
    const int m = lane & 15, half = lane >> 4;
    const int xc = px0 + m;
    const __bf16* xb = xbf + (size_t)b * HW * 64;
    const float* ob = off + (size_t)b * 18 * HW + y * WW + xc;

    v8f acc[4] = {};
    int k = 0;
    for (int ki = -1; ki <= 1; ++ki) {
        for (int kj = -1; kj <= 1; ++kj, ++k) {
            float ody = ob[(size_t)(2 * k)     * HW];
            float odx = ob[(size_t)(2 * k + 1) * HW];
            float sy = (float)(y + ki)  + ody;
            float sx = (float)(xc + kj) + odx;
            float fy = floorf(sy), fx = floorf(sx);
            float wy1 = sy - fy, wy0 = 1.f - wy1;
            float wx1 = sx - fx, wx0 = 1.f - wx1;
            int y0 = (int)fy, x0 = (int)fx, y1 = y0 + 1, x1 = x0 + 1;
            bool vy0 = (unsigned)y0 < 128u, vy1 = (unsigned)y1 < 128u;
            bool vx0 = (unsigned)x0 < 128u, vx1 = (unsigned)x1 < 128u;
            float w00 = wy0 * wx0 * (float)(vy0 && vx0);
            float w01 = wy0 * wx1 * (float)(vy0 && vx1);
            float w10 = wy1 * wx0 * (float)(vy1 && vx0);
            float w11 = wy1 * wx1 * (float)(vy1 && vx1);
            const int ry0 = iclamp(y0, 0, 127) * WW, ry1 = iclamp(y1, 0, 127) * WW;
            const int rx0 = iclamp(x0, 0, 127),      rx1 = iclamp(x1, 0, 127);
            const __bf16* p00 = xb + (size_t)(ry0 + rx0) * 64;
            const __bf16* p01 = xb + (size_t)(ry0 + rx1) * 64;
            const __bf16* p10 = xb + (size_t)(ry1 + rx0) * 64;
            const __bf16* p11 = xb + (size_t)(ry1 + rx1) * 64;

#pragma unroll
            for (int cc = 0; cc < 64; cc += 32) {
                union { unsigned u[8]; v16bf v; } af;
#pragma unroll
                for (int r = 0; r < 2; ++r) {
                    int cb = cc + half * 8 + r * 16;
                    v4u q00 = *(const v4u*)(p00 + cb);
                    v4u q01 = *(const v4u*)(p01 + cb);
                    v4u q10 = *(const v4u*)(p10 + cb);
                    v4u q11 = *(const v4u*)(p11 + cb);
#pragma unroll
                    for (int j = 0; j < 4; ++j) {
                        float gl = w00 * bflo(q00[j]) + w01 * bflo(q01[j]) +
                                   w10 * bflo(q10[j]) + w11 * bflo(q11[j]);
                        float gh = w00 * bfhi(q00[j]) + w01 * bfhi(q01[j]) +
                                   w10 * bfhi(q10[j]) + w11 * bfhi(q11[j]);
                        af.u[r * 4 + j] = packbf2(gl, gh);
                    }
                }
                const int chunk = k * 2 + (cc >> 5);
#pragma unroll
                for (int nf = 0; nf < 4; ++nf)
                    acc[nf] = wmma_bf16(af.v, load_bfrag(Wd, chunk, 4, nf, lane), acc[nf]);
            }
        }
    }
#pragma unroll
    for (int nf = 0; nf < 4; ++nf) {
        int co = nf * 16 + m;
        float bi = dbias[co];
        float* op = Fd + ((size_t)b * NC + co) * HW + y * WW + px0 + half * 8;
#pragma unroll
        for (int i = 0; i < 8; ++i) op[i] = acc[nf][i] + bi;
    }
}

// ============ softmax fusion: Fs,Fd f32 NCHW -> fused bf16 NHWC ============
__global__ void k_fuse(const float* __restrict__ Fs, const float* __restrict__ Fd,
                       const float* __restrict__ wg_w, const float* __restrict__ wg_b,
                       __bf16* __restrict__ Fb) {
    int p = blockIdx.x * 256 + threadIdx.x;             // 131072
    int xc = p & 127, y = (p >> 7) & 127, b = p >> 14;
    size_t base = (size_t)b * NC * HW + y * WW + xc;
    float l0 = wg_b[0], l1 = wg_b[1];
    for (int c = 0; c < NC; ++c) {
        float fs = Fs[base + (size_t)c * HW];
        float fd = Fd[base + (size_t)c * HW];
        l0 += wg_w[c] * fs + wg_w[64 + c] * fd;
        l1 += wg_w[128 + c] * fs + wg_w[192 + c] * fd;
    }
    float mx = fmaxf(l0, l1);
    float e0 = __expf(l0 - mx), e1 = __expf(l1 - mx);
    float w0 = e0 / (e0 + e1), w1 = 1.f - w0;
    __bf16* fo = Fb + (size_t)p * 64;
    for (int c = 0; c < NC; ++c)
        fo[c] = f2bf(Fs[base + (size_t)c * HW] * w0 + Fd[base + (size_t)c * HW] * w1);
}

// ============ folded ELK: 13-tap conv (WMMA), writes final output ============
__global__ void k_elk(const __bf16* __restrict__ Fb, const __bf16* __restrict__ We,
                      const float* __restrict__ be, float* __restrict__ out) {
    const int tile = blockIdx.x, lane = threadIdx.x;    // 8192 x 32
    const int px0 = (tile & 7) * 16, y = (tile >> 3) & 127, b = tile >> 10;
    const int m = lane & 15, half = lane >> 4;
    const int xc = px0 + m;
    const __bf16* fb = Fb + (size_t)b * HW * 64;

    v8f acc[4] = {};
    for (int t = 0; t < 13; ++t) {
        const int yy = y + cDY[t], xx = xc + cDX[t];
        const unsigned msk = ((unsigned)yy < 128u && (unsigned)xx < 128u) ? ~0u : 0u;
        const __bf16* pix = fb + (size_t)(iclamp(yy, 0, 127) * WW + iclamp(xx, 0, 127)) * 64;
        __builtin_prefetch(pix + 64 * WW, 0, 1);
#pragma unroll
        for (int cc = 0; cc < 64; cc += 32) {
            v16bf a = load_afrag(pix, cc, half, msk);
            const int chunk = t * 2 + (cc >> 5);
#pragma unroll
            for (int nf = 0; nf < 4; ++nf)
                acc[nf] = wmma_bf16(a, load_bfrag(We, chunk, 4, nf, lane), acc[nf]);
        }
    }
#pragma unroll
    for (int nf = 0; nf < 4; ++nf) {
        int co = nf * 16 + m;
        float bi = be[co];
        float* op = out + ((size_t)b * NC + co) * HW + y * WW + px0 + half * 8;
#pragma unroll
        for (int i = 0; i < 8; ++i) op[i] = acc[nf][i] + bi;
    }
}

// ---------------- host orchestration ----------------
extern "C" void kernel_launch(void* const* d_in, const int* in_sizes, int n_in,
                              void* d_out, int out_size, void* d_ws, size_t ws_size,
                              hipStream_t stream) {
    (void)in_sizes; (void)n_in; (void)out_size; (void)ws_size;
    const float* x      = (const float*)d_in[0];
    const float* offset = (const float*)d_in[1];
    const float* stn_w1 = (const float*)d_in[2];
    const float* stn_b1 = (const float*)d_in[3];
    const float* stn_w2 = (const float*)d_in[4];
    const float* stn_b2 = (const float*)d_in[5];
    const float* fc_w   = (const float*)d_in[6];
    const float* fc_b   = (const float*)d_in[7];
    const float* dcn_w  = (const float*)d_in[8];
    const float* dcn_b  = (const float*)d_in[9];
    const float* wg_w   = (const float*)d_in[10];
    const float* wg_b   = (const float*)d_in[11];
    const float* w3     = (const float*)d_in[12];
    const float* b3     = (const float*)d_in[13];
    const float* w15    = (const float*)d_in[14];
    const float* b15    = (const float*)d_in[15];
    const float* w51    = (const float*)d_in[16];
    const float* b51    = (const float*)d_in[17];
    const float* w1     = (const float*)d_in[18];
    const float* b1     = (const float*)d_in[19];
    float* out = (float*)d_out;

    char* ws = (char*)d_ws;
    size_t o = 0;
    __bf16* xbf  = (__bf16*)(ws + o); o += (size_t)8 * HW * 64 * 2;       // 16.8 MB
    float*  h1   = (float*)(ws + o);  o += (size_t)8 * 32 * HW * 4;       // 16.8 MB (reused as Fb)
    __bf16* Fb   = (__bf16*)h1;                                           // fused bf16 NHWC (h1 dead by then)
    __bf16* h1pb = (__bf16*)(ws + o); o += (size_t)8 * 4096 * 32 * 2;     //  2.1 MB
    float*  h2   = (float*)(ws + o);  o += (size_t)8 * 64 * 4096 * 4;     //  8.4 MB
    float*  feat = (float*)(ws + o);  o += 2048;
    float*  theta= (float*)(ws + o);  o += 256;
    float*  Fs   = (float*)(ws + o);  o += (size_t)8 * NC * HW * 4;       // 33.5 MB
    float*  Fd   = (float*)(ws + o);  o += (size_t)8 * NC * HW * 4;       // 33.5 MB
    __bf16* W1b  = (__bf16*)(ws + o); o += (size_t)100352 * 2;
    __bf16* W2b  = (__bf16*)(ws + o); o += (size_t)51200 * 2;
    __bf16* Wdb  = (__bf16*)(ws + o); o += (size_t)36864 * 2;
    __bf16* Web  = (__bf16*)(ws + o); o += (size_t)53248 * 2;
    float*  beff = (float*)(ws + o);  o += 256;

    // weight prep + activation layout conversion
    k_cvt_stn1<<<(100352 + 255) / 256, 256, 0, stream>>>(stn_w1, W1b);
    k_cvt_stn2<<<(51200 + 255) / 256, 256, 0, stream>>>(stn_w2, W2b);
    k_cvt_dcn <<<(36864 + 255) / 256, 256, 0, stream>>>(dcn_w, Wdb);
    k_elk_fold<<<(53248 + 64 + 255) / 256, 256, 0, stream>>>(
        w3, w15, w51, w1, b3, b15, b51, b1, Web, beff);
    k_x2bf<<<2048, 256, 0, stream>>>(x, xbf);

    // STN localization
    k_stn_conv1<<<8192, 32, 0, stream>>>(xbf, W1b, stn_b1, h1);
    k_pool_cvt <<<512, 256, 0, stream>>>(h1, h1pb);
    k_stn_conv2<<<2048, 32, 0, stream>>>(h1pb, W2b, stn_b2, h2);
    k_gap      <<<512, 256, 0, stream>>>(h2, feat);
    k_fc       <<<1, 64, 0, stream>>>(feat, fc_w, fc_b, theta);

    // branches
    k_sample<<<512, 256, 0, stream>>>(xbf, theta, Fs);
    k_dcn   <<<8192, 32, 0, stream>>>(xbf, offset, Wdb, dcn_b, Fd);

    // fusion (writes bf16 NHWC into dead h1 region) + folded ELK
    k_fuse<<<512, 256, 0, stream>>>(Fs, Fd, wg_w, wg_b, Fb);
    k_elk <<<8192, 32, 0, stream>>>(Fb, Web, beff, out);
}
